// MGSGCN_21964462751838
// MI455X (gfx1250) — compile-verified, matrix operand order checked by reading, not run on hardware
//
#include <hip/hip_runtime.h>
#include <hip/hip_bf16.h>
#include <string.h>

typedef __bf16 bf16;
typedef __attribute__((ext_vector_type(16))) __bf16 v16bf;
typedef __attribute__((ext_vector_type(8)))  float  v8f;

static constexpr int NN   = 50000;    // nodes
static constexpr int NE   = 800000;   // edges (before self loops)
static constexpr int NP   = 1024;     // pairs
static constexpr int FIN  = 128;      // input feature dim
static constexpr int HD   = 256;      // HEADS*CH
static constexpr int NT   = NN / 16;  // 3125 row tiles (exact)

// ---------------------------------------------------------------------------
// WMMA fragment swizzle (CDNA5 16-bit A/B layout, wave32).
// A tile: 16 rows x 32 K.  B tile: 32 K x 16 cols.  Stored as [lane][16 elems].
// For K-in-tile k: group=k>>4, half=(k&15)>>3, vi=(k>>1)&3, b=k&1
//   lane = half*16 + (row|col),  elem = group*8 + vi*2 + b
// ---------------------------------------------------------------------------
__device__ __forceinline__ int swz_index(int rc, int k) {
    int group = k >> 4;
    int half  = (k & 15) >> 3;
    int vi    = (k >> 1) & 3;
    int b     = k & 1;
    return (half * 16 + rc) * 16 + (group * 8 + vi * 2 + b);
}

// fragment load: one contiguous 32B v16bf per lane
__device__ __forceinline__ v16bf load_frag(const bf16* base, int tile) {
    return *((const v16bf*)(base + (size_t)tile * 512) + (threadIdx.x & 31));
}

// f32 C/D store (row-major): lane -> col = lane&15, rows v + 8*(lane>>4)
__device__ __forceinline__ void store_c_frag(float* base, v8f c, int row0, int col0, int ld) {
    int lane = threadIdx.x & 31;
    int col  = col0 + (lane & 15);
    int half = lane >> 4;
#pragma unroll
    for (int v = 0; v < 8; ++v)
        base[(size_t)(row0 + half * 8 + v) * ld + col] = c[v];
}

// ---------------------------------------------------------------------------
// Utility kernels
// ---------------------------------------------------------------------------
__global__ void fill_u32(unsigned* p, unsigned v, int n) {
    int i = blockIdx.x * blockDim.x + threadIdx.x;
    if (i < n) p[i] = v;
}

// x [N,128] f32 -> A-swizzled bf16 tiles [NT][4][512]
__global__ void cvt_x_swz(const float* s, bf16* d, int n) {
    int i = blockIdx.x * blockDim.x + threadIdx.x;
    if (i >= n) return;
    int nd = i >> 7, k = i & 127;
    int tile = (nd >> 4) * 4 + (k >> 5);
    d[(size_t)tile * 512 + swz_index(nd & 15, k & 31)] = (bf16)s[i];
}

// W [kdim? x 256] f32 -> B-swizzled bf16 tiles [kdim/32][16][512]; zero-pad rows>=rows
__global__ void pack_w_swz(const float* s, bf16* d, int rows, int padRows) {
    int i = blockIdx.x * blockDim.x + threadIdx.x;
    if (i >= padRows * HD) return;
    int k = i / HD, c = i % HD;
    float v = (k < rows) ? s[k * HD + c] : 0.0f;
    int tile = (k >> 5) * 16 + (c >> 4);
    d[(size_t)tile * 512 + swz_index(c & 15, k & 31)] = (bf16)v;
}

// ---------------------------------------------------------------------------
// Dual-path precompute: M = (weight_d @ H_d_base) @ weight_right  [128,16]
// ---------------------------------------------------------------------------
__global__ void dual_t1(const float* wd, const float* hd, float* t1) {
    int i = blockIdx.x * blockDim.x + threadIdx.x;
    if (i >= 128 * 512) return;
    int r = i / 512, c = i % 512;
    float s = 0.f;
    for (int k = 0; k < 512; ++k) s += wd[r * 512 + k] * hd[k * 512 + c];
    t1[i] = s;
}

__global__ void dual_m(const float* t1, const float* wr, float* m) {
    int i = blockIdx.x * blockDim.x + threadIdx.x;
    if (i >= 128 * 16) return;
    int r = i / 16, c = i % 16;
    float s = 0.f;
    for (int k = 0; k < 512; ++k) s += t1[r * 512 + k] * wr[k * 16 + c];
    m[i] = s;
}

// Wcat = [W_gcn | M] (128 x 32) -> B-swizzled tiles [4 ktiles][2 coltiles][512]
__global__ void pack_wcat_swz(const float* wgcn, const float* m, bf16* d) {
    int i = blockIdx.x * blockDim.x + threadIdx.x;
    if (i >= 128 * 32) return;
    int k = i / 32, c = i % 32;
    float v = (c < 16) ? wgcn[k * 16 + c] : m[k * 16 + (c - 16)];
    int tile = (k >> 5) * 2 + (c >> 4);
    d[(size_t)tile * 512 + swz_index(c & 15, k & 31)] = (bf16)v;
}

// ---------------------------------------------------------------------------
// Fused node GEMM: [xw_gcn | dual] = x @ [W_gcn | M]   (one wave per 16 rows)
// ---------------------------------------------------------------------------
__global__ __launch_bounds__(32) void node_gemm_xw(const bf16* xb, const bf16* wcat,
                                                   float* xwg, float* dual) {
    int rt = blockIdx.x;
    v8f c0 = {}, c1 = {};
#pragma unroll
    for (int kt = 0; kt < 4; ++kt) {
        v16bf a  = load_frag(xb,   rt * 4 + kt);
        v16bf b0 = load_frag(wcat, kt * 2 + 0);
        v16bf b1 = load_frag(wcat, kt * 2 + 1);
        c0 = __builtin_amdgcn_wmma_f32_16x16x32_bf16(false, a, false, b0, (short)0, c0, false, false);
        c1 = __builtin_amdgcn_wmma_f32_16x16x32_bf16(false, a, false, b1, (short)0, c1, false, false);
    }
    store_c_frag(xwg,  c0, rt * 16, 0, 16);
    store_c_frag(dual, c1, rt * 16, 0, 16);
}

// ---------------------------------------------------------------------------
// GCN aggregation
// ---------------------------------------------------------------------------
__global__ void deg_count(const int* ei, float* deg) {
    int e = blockIdx.x * blockDim.x + threadIdx.x;
    if (e < NE) atomicAdd(&deg[ei[NE + e]], 1.0f);
}

__global__ void gcn_agg(const int* ei, const float* deg, const float* xwg, float* conv1) {
    int i = blockIdx.x * blockDim.x + threadIdx.x;
    if (i >= NE * 16) return;
    int e = i >> 4, c = i & 15;
    int s = ei[e], d = ei[NE + e];
    float norm = rsqrtf(deg[s]) * rsqrtf(deg[d]);
    atomicAdd(&conv1[d * 16 + c], norm * xwg[s * 16 + c]);
}

// h0 = elu(conv1 + selfloop + dual + b_gcn) -> A-swizzled bf16 [NT][1][512] (K padded to 32)
__global__ void h0_build(const float* conv1, const float* xwg, const float* dual,
                         const float* deg, const float* bgcn, bf16* hb) {
    int i = blockIdx.x * blockDim.x + threadIdx.x;
    if (i >= NN * 32) return;
    int n = i >> 5, c = i & 31;
    float e = 0.f;
    if (c < 16) {
        float s = conv1[n * 16 + c] + xwg[n * 16 + c] / deg[n] + dual[n * 16 + c] + bgcn[c];
        e = s > 0.f ? s : __expf(s) - 1.f;
    }
    hb[(size_t)(n >> 4) * 512 + swz_index(n & 15, c)] = (bf16)e;
}

// ---------------------------------------------------------------------------
// GAT layer GEMM: hW = hb @ Wb, one wave -> 16 rows x 64 cols
//   hb: A-swizzled [NT][ktiles][512], wb: B-swizzled [ktiles][16][512]
// ---------------------------------------------------------------------------
__global__ __launch_bounds__(32) void gat_gemm(const bf16* hb, const bf16* wb,
                                               float* hw, int ktiles) {
    int rt  = blockIdx.x;
    int ct0 = blockIdx.y * 4;
    v8f acc[4] = {{}, {}, {}, {}};
    for (int kt = 0; kt < ktiles; ++kt) {
        v16bf a = load_frag(hb, rt * ktiles + kt);
#pragma unroll
        for (int t = 0; t < 4; ++t) {
            v16bf b = load_frag(wb, kt * 16 + ct0 + t);
            acc[t] = __builtin_amdgcn_wmma_f32_16x16x32_bf16(false, a, false, b, (short)0,
                                                             acc[t], false, false);
        }
    }
#pragma unroll
    for (int t = 0; t < 4; ++t) store_c_frag(hw, acc[t], rt * 16, (ct0 + t) * 16, HD);
}

// per-node attention coefficients sa/da [N,16]
__global__ void attn_coef(const float* hw, const float* asrc, const float* adst,
                          float* sa, float* da) {
    int i = blockIdx.x * blockDim.x + threadIdx.x;
    if (i >= NN * 16) return;
    int n = i >> 4, h = i & 15;
    float s = 0.f, d = 0.f;
    for (int c = 0; c < 16; ++c) {
        float v = hw[(size_t)n * HD + h * 16 + c];
        s += v * asrc[h * 16 + c];
        d += v * adst[h * 16 + c];
    }
    sa[i] = s;
    da[i] = d;
}

// order-preserving float<->uint for atomicMax
__device__ __forceinline__ unsigned fkey(float x) {
    unsigned b = __float_as_uint(x);
    return (b & 0x80000000u) ? ~b : (b | 0x80000000u);
}
__device__ __forceinline__ float fdec(unsigned k) {
    return (k & 0x80000000u) ? __uint_as_float(k & 0x7FFFFFFFu) : __uint_as_float(~k);
}

__device__ __forceinline__ void edge_sd(const int* ei, int i, int& s, int& d) {
    if (i < NE) { s = ei[i]; d = ei[NE + i]; }
    else        { s = d = i - NE; }           // self loop
}
__device__ __forceinline__ float lrelu02(float v) { return v > 0.f ? v : 0.2f * v; }

__global__ void edge_max(const int* ei, const float* sa, const float* da, unsigned* menc) {
    int i = blockIdx.x * blockDim.x + threadIdx.x;
    if (i >= (NE + NN) * 16) return;
    int e = i >> 4, h = i & 15, s, d;
    edge_sd(ei, e, s, d);
    float l = lrelu02(sa[s * 16 + h] + da[d * 16 + h]);
    atomicMax(&menc[d * 16 + h], fkey(l));
}

__global__ void edge_expsum(const int* ei, const float* sa, const float* da,
                            const unsigned* menc, float* z) {
    int i = blockIdx.x * blockDim.x + threadIdx.x;
    if (i >= (NE + NN) * 16) return;
    int e = i >> 4, h = i & 15, s, d;
    edge_sd(ei, e, s, d);
    float l = lrelu02(sa[s * 16 + h] + da[d * 16 + h]);
    atomicAdd(&z[d * 16 + h], __expf(l - fdec(menc[d * 16 + h])));
}

__global__ void edge_agg(const int* ei, const float* sa, const float* da,
                         const unsigned* menc, const float* z, const float* hw, float* out) {
    int i = blockIdx.x * blockDim.x + threadIdx.x;
    if (i >= (NE + NN) * 16) return;
    int e = i >> 4, h = i & 15, s, d;
    edge_sd(ei, e, s, d);
    float l = lrelu02(sa[s * 16 + h] + da[d * 16 + h]);
    float alpha = __expf(l - fdec(menc[d * 16 + h])) / (z[d * 16 + h] + 1e-16f);
    const float* hs = hw + (size_t)s * HD + h * 16;
    float* od = out + (size_t)d * HD + h * 16;
    for (int c = 0; c < 16; ++c) atomicAdd(&od[c], alpha * hs[c]);
}

// out = elu(out + b); f32 in place + A-swizzled bf16 [NT][8][512] for next GEMM
__global__ void gat_finish(float* x, const float* b, bf16* hb) {
    int i = blockIdx.x * blockDim.x + threadIdx.x;
    if (i >= NN * HD) return;
    int n = i >> 8, k = i & 255;
    float v = x[i] + b[k];
    float e = v > 0.f ? v : __expf(v) - 1.f;
    x[i] = e;
    hb[(size_t)((n >> 4) * 8 + (k >> 5)) * 512 + swz_index(n & 15, k & 31)] = (bf16)e;
}

// ---------------------------------------------------------------------------
// Final DNN + log_softmax (one wave per pair)
// ---------------------------------------------------------------------------
__global__ __launch_bounds__(32) void final_dnn(const float* h, const int* uidx, const int* iidx,
                                                const float* wdnn, float* out) {
    int p = blockIdx.x;
    int lane = threadIdx.x;
    int u = uidx[p], it = iidx[p];
    float s0 = 0.f, s1 = 0.f;
    for (int k = lane; k < 2 * HD; k += 32) {
        float f = (k < HD) ? h[(size_t)u * HD + k] : h[(size_t)it * HD + (k - HD)];
        s0 += f * wdnn[k * 2 + 0];
        s1 += f * wdnn[k * 2 + 1];
    }
    for (int o = 16; o > 0; o >>= 1) {
        s0 += __shfl_xor(s0, o, 32);
        s1 += __shfl_xor(s1, o, 32);
    }
    if (lane == 0) {
        float m   = fmaxf(s0, s1);
        float lse = m + __logf(__expf(s0 - m) + __expf(s1 - m));
        out[p * 2 + 0] = s0 - lse;
        out[p * 2 + 1] = s1 - lse;
    }
}

// ---------------------------------------------------------------------------
// Host launch
// ---------------------------------------------------------------------------
static inline size_t alignup(size_t x) { return (x + 255) & ~size_t(255); }

extern "C" void kernel_launch(void* const* d_in, const int* in_sizes, int n_in,
                              void* d_out, int out_size, void* d_ws, size_t ws_size,
                              hipStream_t stream) {
    const float* x    = (const float*)d_in[0];
    const int*   ei   = (const int*)d_in[1];
    const float* hd   = (const float*)d_in[2];
    const int*   uidx = (const int*)d_in[3];
    const int*   iidx = (const int*)d_in[4];
    const float* Wgcn = (const float*)d_in[5];
    const float* bgcn = (const float*)d_in[6];
    const float* W[3]    = {(const float*)d_in[7],  (const float*)d_in[11], (const float*)d_in[15]};
    const float* asrc[3] = {(const float*)d_in[8],  (const float*)d_in[12], (const float*)d_in[16]};
    const float* adst[3] = {(const float*)d_in[9],  (const float*)d_in[13], (const float*)d_in[17]};
    const float* bb[3]   = {(const float*)d_in[10], (const float*)d_in[14], (const float*)d_in[18]};
    const float* wd   = (const float*)d_in[19];
    const float* wr   = (const float*)d_in[20];
    const float* wdnn = (const float*)d_in[21];
    float* out = (float*)d_out;

    // workspace carve-out
    char* ws = (char*)d_ws;
    size_t cur = 0;
    auto alloc = [&](size_t bytes) { char* p = ws + cur; cur += alignup(bytes); return p; };
    bf16*     xb    = (bf16*)alloc((size_t)NN * FIN * 2);   // A-swizzled x
    float*    t1    = (float*)alloc(128 * 512 * 4);
    float*    mf    = (float*)alloc(128 * 16 * 4);
    bf16*     wcat  = (bf16*)alloc(128 * 32 * 2);           // B-swizzled [Wgcn|M]
    bf16*     wb    = (bf16*)alloc(HD * HD * 2);            // B-swizzled layer weight
    float*    xwg   = (float*)alloc((size_t)NN * 16 * 4);
    float*    dual  = (float*)alloc((size_t)NN * 16 * 4);
    float*    deg   = (float*)alloc((size_t)NN * 4);
    float*    conv1 = (float*)alloc((size_t)NN * 16 * 4);
    float*    sa    = (float*)alloc((size_t)NN * 16 * 4);
    float*    da    = (float*)alloc((size_t)NN * 16 * 4);
    unsigned* menc  = (unsigned*)alloc((size_t)NN * 16 * 4);
    float*    z     = (float*)alloc((size_t)NN * 16 * 4);
    bf16*     hb    = (bf16*)alloc((size_t)NN * HD * 2);    // A-swizzled activations
    float*    X     = (float*)alloc((size_t)NN * HD * 4);   // agg accum / layer out f32
    float*    Y     = (float*)alloc((size_t)NN * HD * 4);   // hW

    const int B = 256;
    auto blocks = [&](long n) { return (int)((n + B - 1) / B); };
    unsigned ONE_F; { float f = 1.0f; memcpy(&ONE_F, &f, 4); }

    // 1. convert x to A-swizzled bf16
    cvt_x_swz<<<blocks((long)NN * FIN), B, 0, stream>>>(x, xb, NN * FIN);

    // 2. dual precompute M = (wd @ hd) @ wr, fuse into Wcat = [Wgcn | M]
    dual_t1<<<blocks(128 * 512), B, 0, stream>>>(wd, hd, t1);
    dual_m<<<blocks(128 * 16), B, 0, stream>>>(t1, wr, mf);
    pack_wcat_swz<<<blocks(128 * 32), B, 0, stream>>>(Wgcn, mf, wcat);

    // 3. fused node GEMM (WMMA): [xw_gcn | dual] = x @ Wcat
    node_gemm_xw<<<NT, 32, 0, stream>>>(xb, wcat, xwg, dual);

    // 4. GCN: degrees (self loop = init 1.0), normalized aggregation
    fill_u32<<<blocks(NN), B, 0, stream>>>((unsigned*)deg, ONE_F, NN);
    deg_count<<<blocks(NE), B, 0, stream>>>(ei, deg);
    fill_u32<<<blocks(NN * 16), B, 0, stream>>>((unsigned*)conv1, 0u, NN * 16);
    gcn_agg<<<blocks((long)NE * 16), B, 0, stream>>>(ei, deg, xwg, conv1);
    h0_build<<<blocks((long)NN * 32), B, 0, stream>>>(conv1, xwg, dual, deg, bgcn, hb);

    // 5. three GAT layers
    for (int l = 0; l < 3; ++l) {
        int ktiles = (l == 0) ? 1 : 8;       // K = 32 (padded) or 256
        int wrows  = (l == 0) ? 16 : HD;
        pack_w_swz<<<blocks((long)ktiles * 32 * HD), B, 0, stream>>>(W[l], wb, wrows, ktiles * 32);
        gat_gemm<<<dim3(NT, 4), 32, 0, stream>>>(hb, wb, Y, ktiles);
        attn_coef<<<blocks(NN * 16), B, 0, stream>>>(Y, asrc[l], adst[l], sa, da);
        fill_u32<<<blocks(NN * 16), B, 0, stream>>>(menc, 0u, NN * 16);
        fill_u32<<<blocks(NN * 16), B, 0, stream>>>((unsigned*)z, 0u, NN * 16);
        fill_u32<<<blocks((long)NN * HD), B, 0, stream>>>((unsigned*)X, 0u, NN * HD);
        long ne16 = (long)(NE + NN) * 16;
        edge_max<<<blocks(ne16), B, 0, stream>>>(ei, sa, da, menc);
        edge_expsum<<<blocks(ne16), B, 0, stream>>>(ei, sa, da, menc, z);
        edge_agg<<<blocks(ne16), B, 0, stream>>>(ei, sa, da, menc, z, Y, X);
        gat_finish<<<blocks((long)NN * HD), B, 0, stream>>>(X, bb[l], hb);
    }

    // 6. final DNN + log_softmax
    final_dnn<<<NP, 32, 0, stream>>>(X, uidx, iidx, wdnn, out);
    (void)in_sizes; (void)n_in; (void)out_size; (void)ws_size;
}
